// ALIGNN4inverse_19456201850984
// MI455X (gfx1250) — compile-verified, matrix operand order checked by reading, not run on hardware
//
#include <hip/hip_runtime.h>

// ---------------------------------------------------------------------------
// ALIGNN forward for MI455X (gfx1250).
// All linears run through v_wmma_f32_16x16x32_f16 with compile-time shapes:
// LDS is staged in WMMA *fragment order* so each lane loads its v16h operand
// with two ds_load_b128 (no scalar u16 gather), and the K loop fully unrolls.
// ---------------------------------------------------------------------------

#define HD 128
#define NN_NODES 8192
#define NE_EDGES 131072
#define NT_TRIP  262144
#define NG_GRAPH 64
#define ATOMF    92
#define EBINS    80
#define ABINS    40
#define EMBD     64

typedef __attribute__((ext_vector_type(16))) _Float16 v16h;
typedef __attribute__((ext_vector_type(8)))  float    v8f;

__device__ __forceinline__ void atomicAddF(float* p, float v) {
  unsafeAtomicAdd(p, v);   // hw global_atomic_add_f32 on gfx1250
}

__device__ __forceinline__ unsigned pack_f16x2(float a, float b) {
  unsigned short ua = __builtin_bit_cast(unsigned short, (_Float16)a);
  unsigned short ub = __builtin_bit_cast(unsigned short, (_Float16)b);
  return (unsigned)ua | ((unsigned)ub << 16);
}

// Fragment-order helpers (ISA 7.12.2, 16-bit operands, wave32):
// For a 16x32 k-chunk, lane = lhalf*16 + lrow, and half-index h within the
// lane's 16 contiguous f16 values maps k-in-chunk kin = g*16 + lhalf*8 + p*2 + q
// to h = 8*g + 2*p + q.
__device__ __forceinline__ int frag_lane(int lrow, int kin) {
  return (((kin >> 3) & 1) << 4) + lrow;
}
__device__ __forceinline__ int frag_half(int kin) {
  return ((kin >> 4) << 3) + (((kin >> 1) & 3) << 1) + (kin & 1);
}

// ---------------------------------------------------------------------------
// C[M,N] = A[M,K] @ W[K,N] + bias[N] (+ addend[M,N] if non-null).
// Compile-time K, N; MT rows per block; blockDim = 32*(N/16) (wave32).
// ---------------------------------------------------------------------------
template <int K, int N, int MT>
__global__ __launch_bounds__(256) void wmma_linear_kernel(
    const float* __restrict__ A, const float* __restrict__ W,
    const float* __restrict__ bias, const float* __restrict__ addend,
    float* __restrict__ C, int M)
{
  constexpr int KP    = (K + 31) & ~31;   // K padded to 32
  constexpr int NK    = KP / 32;          // k-chunks
  constexpr int NSUB  = MT / 16;          // 16-row sub-tiles per block
  constexpr int NTILE = N / 16;           // 16-col tiles (= waves per block)

  // LDS already laid out in fragment order: [chunk][tile][lane][16 halves]
  __shared__ __align__(32) _Float16 sA[NK * NSUB * 32 * 16];
  __shared__ __align__(32) _Float16 sW[NK * NTILE * 32 * 16];

  const int tid  = threadIdx.x;
  const int nthr = blockDim.x;
  const int rBase = blockIdx.x * MT;

  // ---- Stage weights fp32 -> f16 pairs in fragment order ----
  // i enumerates (k-pair, n); consecutive threads touch consecutive n (coalesced).
  for (int i = tid; i < (KP / 2) * N; i += nthr) {
    int kp = i / N, n = i - kp * N;
    int k2 = kp * 2;
    float v0 = (k2 < K)     ? W[k2 * N + n]       : 0.0f;
    float v1 = (k2 + 1 < K) ? W[(k2 + 1) * N + n] : 0.0f;
    int kc = k2 >> 5, kin = k2 & 31;
    int lane = frag_lane(n & 15, kin);
    int h    = frag_half(kin);                 // even (q=0)
    int idx = (((kc * NTILE + (n >> 4)) * 32 + lane) << 4) + h;
    *reinterpret_cast<unsigned*>(&sW[idx]) = pack_f16x2(v0, v1);
  }
  // ---- Stage A tile fp32 -> f16 pairs in fragment order ----
  // i enumerates (row, k-pair); consecutive threads load 8B-contiguous A.
  for (int i = tid; i < MT * (KP / 2); i += nthr) {
    int r = i / (KP / 2), kp = i - r * (KP / 2);
    int k2 = kp * 2;
    int row = rBase + r;
    bool rok = row < M;
    float v0 = (rok && k2 < K)     ? A[row * K + k2]     : 0.0f;
    float v1 = (rok && k2 + 1 < K) ? A[row * K + k2 + 1] : 0.0f;
    int kc = k2 >> 5, kin = k2 & 31;
    int lane = frag_lane(r & 15, kin);
    int h    = frag_half(kin);
    int idx = (((kc * NSUB + (r >> 4)) * 32 + lane) << 4) + h;
    *reinterpret_cast<unsigned*>(&sA[idx]) = pack_f16x2(v0, v1);
  }
  __syncthreads();

  const int wave  = tid >> 5;      // = n-tile
  const int lane  = tid & 31;
  const int nBase = wave * 16;
  const int lrow  = lane & 15;
  const int lhalf = lane >> 4;

  v8f acc[NSUB];
#pragma unroll
  for (int ms = 0; ms < NSUB; ++ms) acc[ms] = (v8f){};

#pragma unroll
  for (int kc = 0; kc < NK; ++kc) {
    const v16h bfrag =
        *reinterpret_cast<const v16h*>(&sW[(((kc * NTILE + wave) * 32 + lane) << 4)]);
#pragma unroll
    for (int ms = 0; ms < NSUB; ++ms) {
      const v16h afrag =
          *reinterpret_cast<const v16h*>(&sA[(((kc * NSUB + ms) * 32 + lane) << 4)]);
      acc[ms] = __builtin_amdgcn_wmma_f32_16x16x32_f16(false, afrag, false, bfrag,
                                                       (short)0, acc[ms], false, false);
    }
  }

  // C/D layout: lane -> col nBase+lrow; VGPR r -> row lhalf*8 + r within sub-tile
  const int col = nBase + lrow;
  const float bv = bias[col];
#pragma unroll
  for (int ms = 0; ms < NSUB; ++ms) {
#pragma unroll
    for (int r = 0; r < 8; ++r) {
      int row = rBase + ms * 16 + lhalf * 8 + r;
      if (row < M) {
        float o = acc[ms][r] + bv;
        if (addend) o += addend[row * N + col];
        C[row * N + col] = o;
      }
    }
  }
}

// Column mean / 1/sqrt(var+eps) over M rows (biased variance, training BN)
__global__ void col_stats_kernel(const float* __restrict__ X, int M, int N,
                                 float* __restrict__ mean, float* __restrict__ istd)
{
  __shared__ float s1[256], s2[256];
  int col = blockIdx.x;
  float a = 0.f, b = 0.f;
  for (int r = threadIdx.x; r < M; r += blockDim.x) {
    float v = X[r * N + col];
    a += v; b += v * v;
  }
  s1[threadIdx.x] = a; s2[threadIdx.x] = b;
  __syncthreads();
  for (int s = blockDim.x >> 1; s > 0; s >>= 1) {
    if (threadIdx.x < (unsigned)s) { s1[threadIdx.x] += s1[threadIdx.x + s];
                                     s2[threadIdx.x] += s2[threadIdx.x + s]; }
    __syncthreads();
  }
  if (threadIdx.x == 0) {
    float mu = s1[0] / (float)M;
    float var = s2[0] / (float)M - mu * mu;
    mean[col] = mu;
    istd[col] = rsqrtf(var + 1e-5f);
  }
}

// out = (res?) + SiLU( (X - mean)*istd*g + b )
__global__ void bn_silu_kernel(const float* __restrict__ X, const float* __restrict__ mean,
                               const float* __restrict__ istd, const float* __restrict__ g,
                               const float* __restrict__ b, const float* __restrict__ res,
                               float* __restrict__ out, int M, int N)
{
  int i = blockIdx.x * blockDim.x + threadIdx.x;
  if (i >= M * N) return;
  int c = i % N;
  float v = (X[i] - mean[c]) * istd[c] * g[c] + b[c];
  float sv = v * (1.0f / (1.0f + __expf(-v)));
  out[i] = (res ? res[i] : 0.0f) + sv;
}

__global__ void rbf_kernel(const float* __restrict__ d, float* __restrict__ out,
                           int M, int bins, float vmin, float vmax)
{
  int i = blockIdx.x * blockDim.x + threadIdx.x;
  if (i >= M * bins) return;
  int r = i / bins, bb = i - r * bins;
  float step = (vmax - vmin) / (float)(bins - 1);
  float cen = vmin + bb * step;
  float gam = 1.0f / step;
  float t = d[r] - cen;
  out[i] = __expf(-gam * t * t);
}

__global__ void fill_kernel(float* __restrict__ p, float v, int n)
{
  int i = blockIdx.x * blockDim.x + threadIdx.x;
  if (i < n) p[i] = v;
}

// P[e, :] += G[idx[e], :]   (float4 vectorized over HD=128 cols)
__global__ void gather_add_kernel(float* __restrict__ P, const float* __restrict__ G,
                                  const int* __restrict__ idx, int nE)
{
  const int per = HD / 4;
  int i = blockIdx.x * blockDim.x + threadIdx.x;
  if (i >= nE * per) return;
  int e = i / per, c4 = i - e * per;
  const float4* g = (const float4*)(G + idx[e] * HD);
  float4* p = (float4*)(P + e * HD);
  float4 a = p[c4]; float4 b = g[c4];
  a.x += b.x; a.y += b.y; a.z += b.z; a.w += b.w;
  p[c4] = a;
}

// s = sigmoid(Mpre[e,c]); num[dst[e],c] += Bh[src[e],c]*s; den[dst[e],c] += s
__global__ void scatter_kernel(const float* __restrict__ Mpre, const float* __restrict__ Bh,
                               const int* __restrict__ src, const int* __restrict__ dst,
                               float* __restrict__ num, float* __restrict__ den, int nE)
{
  int i = blockIdx.x * blockDim.x + threadIdx.x;
  if (i >= nE * HD) return;
  int e = i / HD, c = i - e * HD;
  float s = 1.0f / (1.0f + __expf(-Mpre[i]));
  atomicAddF(&num[dst[e] * HD + c], Bh[src[e] * HD + c] * s);
  atomicAddF(&den[dst[e] * HD + c], s);
}

__global__ void div_kernel(float* __restrict__ num, const float* __restrict__ den, int n)
{
  int i = blockIdx.x * blockDim.x + threadIdx.x;
  if (i < n) num[i] = num[i] / (den[i] + 1e-6f);
}

__global__ void pool_kernel(const float* __restrict__ x, const int* __restrict__ n2g,
                            float* __restrict__ sums, float* __restrict__ cnt, int nN)
{
  int i = blockIdx.x * blockDim.x + threadIdx.x;
  if (i >= nN * HD) return;
  int n = i / HD, c = i - n * HD;
  int g = n2g[n];
  atomicAddF(&sums[g * HD + c], x[i]);
  if (c == 0) atomicAddF(&cnt[g], 1.0f);
}

__global__ void fc_kernel(const float* __restrict__ sums, const float* __restrict__ cnt,
                          const float* __restrict__ W, const float* __restrict__ b,
                          float* __restrict__ out)
{
  __shared__ float red[HD];
  int g = blockIdx.x, c = threadIdx.x;
  float denom = fmaxf(cnt[g], 1.0f);
  red[c] = (sums[g * HD + c] / denom) * W[c];
  __syncthreads();
  for (int s = HD / 2; s > 0; s >>= 1) {
    if (c < s) red[c] += red[c + s];
    __syncthreads();
  }
  if (c == 0) out[g] = red[0] + b[0];
}

// ---------------------------------------------------------------------------
// Host-side orchestration
// ---------------------------------------------------------------------------
template <int K, int N>
static inline void launch_gemm(hipStream_t s, const float* A, const float* W,
                               const float* bias, const float* add, float* C, int M) {
  constexpr int MT = 64;
  wmma_linear_kernel<K, N, MT>
      <<<dim3((M + MT - 1) / MT), dim3(32 * (N / 16)), 0, s>>>(A, W, bias, add, C, M);
}
static inline void launch_fill(hipStream_t s, float* p, float v, int n) {
  fill_kernel<<<(n + 255) / 256, 256, 0, s>>>(p, v, n);
}
static inline void launch_stats(hipStream_t s, const float* X, int M, int N,
                                float* mean, float* istd) {
  col_stats_kernel<<<N, 256, 0, s>>>(X, M, N, mean, istd);
}
static inline void launch_bnsilu(hipStream_t s, const float* X, const float* mean,
                                 const float* istd, const float* g, const float* b,
                                 const float* res, float* out, int M, int N) {
  int tot = M * N;
  bn_silu_kernel<<<(tot + 255) / 256, 256, 0, s>>>(X, mean, istd, g, b, res, out, M, N);
}

struct EgcnP {
  const float *sgW, *sgb, *dgW, *dgb, *egW, *egb, *suW, *sub, *duW, *dub;
  const float *bnNg, *bnNb, *bnEg, *bnEb;
};
static inline EgcnP egcn_params(void* const* d_in, int b) {
  auto f = [&](int i) { return (const float*)d_in[i]; };
  return { f(b), f(b+1), f(b+2), f(b+3), f(b+4), f(b+5), f(b+6), f(b+7), f(b+8), f(b+9),
           f(b+10), f(b+11), f(b+12), f(b+13) };
}

// EdgeGatedGraphConv: updates x in place, writes y_new to y_out (may alias y_in)
static void run_egcn(hipStream_t s, const EgcnP& p, const int* src, const int* dst,
                     float* x, const float* y_in, float* y_out, int nE, int nN,
                     float* P /*nE x HD*/, float* Q /*nN x HD*/, float* R /*nN x HD*/,
                     float* num /*nN x HD*/, float* den /*nN x HD*/,
                     float* mean, float* istd)
{
  // m = edge_gate(y) + src_gate(x)[src] + dst_gate(x)[dst]
  launch_gemm<HD, HD>(s, y_in, p.egW, p.egb, nullptr, P, nE);
  launch_gemm<HD, HD>(s, x, p.sgW, p.sgb, nullptr, Q, nN);
  gather_add_kernel<<<(nE * (HD / 4) + 255) / 256, 256, 0, s>>>(P, Q, src, nE);
  launch_gemm<HD, HD>(s, x, p.dgW, p.dgb, nullptr, Q, nN);
  gather_add_kernel<<<(nE * (HD / 4) + 255) / 256, 256, 0, s>>>(P, Q, dst, nE);
  // Bh = dst_update(x); gated segment sums at dst
  launch_gemm<HD, HD>(s, x, p.duW, p.dub, nullptr, R, nN);
  launch_fill(s, num, 0.0f, nN * HD);
  launch_fill(s, den, 0.0f, nN * HD);
  scatter_kernel<<<(nE * HD + 255) / 256, 256, 0, s>>>(P, R, src, dst, num, den, nE);
  div_kernel<<<(nN * HD + 255) / 256, 256, 0, s>>>(num, den, nN * HD);   // h
  // t = src_update(x) + h ; x_new = x + SiLU(BN(t))
  launch_gemm<HD, HD>(s, x, p.suW, p.sub, num, Q, nN);
  launch_stats(s, Q, nN, HD, mean, istd);
  launch_bnsilu(s, Q, mean, istd, p.bnNg, p.bnNb, x, x, nN, HD);
  // y_new = y + SiLU(BN(m))
  launch_stats(s, P, nE, HD, mean, istd);
  launch_bnsilu(s, P, mean, istd, p.bnEg, p.bnEb, y_in, y_out, nE, HD);
}

extern "C" void kernel_launch(void* const* d_in, const int* in_sizes, int n_in,
                              void* d_out, int out_size, void* d_ws, size_t ws_size,
                              hipStream_t stream)
{
  (void)in_sizes; (void)n_in; (void)out_size; (void)ws_size;
  auto F = [&](int i) { return (const float*)d_in[i]; };

  // Flat input order (depth-first over setup_inputs() dict / list insertion order):
  // 0 atom_features, 1 bondlength, 2 angle_features,
  // 3..6 atom_emb {W,b,gamma,beta}, 7..10 edge_emb[0], 11..14 edge_emb[1],
  // 15..18 angle_emb[0], 19..22 angle_emb[1],
  // 23..36 alignn[0].node, 37..50 alignn[0].edge, 51..64 alignn[1].node,
  // 65..78 alignn[1].edge, 79..92 gcn[0], 93..106 gcn[1], 107..108 fc {W,b},
  // 109 edge_src, 110 edge_dst, 111 t_src, 112 t_dst, 113 node2graph
  const float* atomF   = F(0);
  const float* bondlen = F(1);
  const float* angleF  = F(2);
  const int* edge_src = (const int*)d_in[109];
  const int* edge_dst = (const int*)d_in[110];
  const int* t_src    = (const int*)d_in[111];
  const int* t_dst    = (const int*)d_in[112];
  const int* n2g      = (const int*)d_in[113];

  // Workspace carve-out (fp32)
  float* ws = (float*)d_ws;
  size_t off = 0;
  auto alloc = [&](size_t n) { float* p = ws + off; off += (n + 63) & ~(size_t)63; return p; };
  float* X   = alloc((size_t)NN_NODES * HD);  // node features
  float* Y   = alloc((size_t)NE_EDGES * HD);  // edge features
  float* Z   = alloc((size_t)NT_TRIP  * HD);  // triplet features
  float* Mb  = alloc((size_t)NE_EDGES * HD);  // m (node-EGCN y_new)
  float* P   = alloc((size_t)NT_TRIP  * HD);  // m_pre / rbf temp
  float* Q   = alloc((size_t)NE_EDGES * HD);  // GEMM temp (nN x HD max)
  float* R   = alloc((size_t)NE_EDGES * HD);  // Bh
  float* num = alloc((size_t)NE_EDGES * HD);
  float* den = alloc((size_t)NE_EDGES * HD);
  float* mean = alloc(HD);
  float* istd = alloc(HD);
  float* psum = alloc((size_t)NG_GRAPH * HD);
  float* pcnt = alloc(NG_GRAPH);

  // ---- Atom embedding: MLP(92 -> 128) ----
  launch_gemm<ATOMF, HD>(stream, atomF, F(3), F(4), nullptr, Q, NN_NODES);
  launch_stats(stream, Q, NN_NODES, HD, mean, istd);
  launch_bnsilu(stream, Q, mean, istd, F(5), F(6), nullptr, X, NN_NODES, HD);

  // ---- Edge embedding: RBF(80) -> MLP(80->64) -> MLP(64->128) ----
  rbf_kernel<<<(NE_EDGES * EBINS + 255) / 256, 256, 0, stream>>>(bondlen, P, NE_EDGES, EBINS, 0.0f, 8.0f);
  launch_gemm<EBINS, EMBD>(stream, P, F(7), F(8), nullptr, Q, NE_EDGES);
  launch_stats(stream, Q, NE_EDGES, EMBD, mean, istd);
  launch_bnsilu(stream, Q, mean, istd, F(9), F(10), nullptr, Q, NE_EDGES, EMBD);
  launch_gemm<EMBD, HD>(stream, Q, F(11), F(12), nullptr, Y, NE_EDGES);
  launch_stats(stream, Y, NE_EDGES, HD, mean, istd);
  launch_bnsilu(stream, Y, mean, istd, F(13), F(14), nullptr, Y, NE_EDGES, HD);

  // ---- Angle embedding: RBF(40) -> MLP(40->64) -> MLP(64->128) ----
  rbf_kernel<<<(NT_TRIP * ABINS + 255) / 256, 256, 0, stream>>>(angleF, P, NT_TRIP, ABINS, -1.0f, 1.0f);
  launch_gemm<ABINS, EMBD>(stream, P, F(15), F(16), nullptr, Q, NT_TRIP);
  launch_stats(stream, Q, NT_TRIP, EMBD, mean, istd);
  launch_bnsilu(stream, Q, mean, istd, F(17), F(18), nullptr, Q, NT_TRIP, EMBD);
  launch_gemm<EMBD, HD>(stream, Q, F(19), F(20), nullptr, Z, NT_TRIP);
  launch_stats(stream, Z, NT_TRIP, HD, mean, istd);
  launch_bnsilu(stream, Z, mean, istd, F(21), F(22), nullptr, Z, NT_TRIP, HD);

  // ---- ALIGNN layers (graph EGCN + line-graph EGCN) ----
  float* yP = Y;
  float* mP = Mb;
  for (int i = 0; i < 2; ++i) {
    int nb = 23 + i * 28;        // node egcn params
    int eb = nb + 14;            // edge (line-graph) egcn params
    EgcnP pn = egcn_params(d_in, nb);
    EgcnP pe = egcn_params(d_in, eb);
    // x, m = egcn(node, g, x, y)
    run_egcn(stream, pn, edge_src, edge_dst, X, yP, mP, NE_EDGES, NN_NODES,
             P, Q, R, num, den, mean, istd);
    // y, z = egcn(edge, lg, m, z)   (m updated in place -> new y; z in place)
    run_egcn(stream, pe, t_src, t_dst, mP, Z, Z, NT_TRIP, NE_EDGES,
             P, Q, R, num, den, mean, istd);
    float* t = yP; yP = mP; mP = t;
  }

  // ---- GCN layers ----
  for (int i = 0; i < 2; ++i) {
    EgcnP pg = egcn_params(d_in, 79 + i * 14);
    run_egcn(stream, pg, edge_src, edge_dst, X, yP, yP, NE_EDGES, NN_NODES,
             P, Q, R, num, den, mean, istd);
  }

  // ---- Average pooling per graph + final FC ----
  launch_fill(stream, psum, 0.0f, NG_GRAPH * HD);
  launch_fill(stream, pcnt, 0.0f, NG_GRAPH);
  pool_kernel<<<(NN_NODES * HD + 255) / 256, 256, 0, stream>>>(X, n2g, psum, pcnt, NN_NODES);
  fc_kernel<<<NG_GRAPH, HD, 0, stream>>>(psum, pcnt, F(107), F(108), (float*)d_out);
}
